// DeepSeekV3Attention_6296422056385
// MI455X (gfx1250) — matrix operand order repaired and checked by reading, not compiled, not run on hardware
//
#include <hip/hip_runtime.h>
#include <hip/hip_bf16.h>

// ---------------------------------------------------------------------------
// DeepSeek-V3 MLA attention for gfx1250 (MI455X), bf16 WMMA throughout.
// B=2, S=2048, E=2048, H=16, NOPE=128, ROPE=64 (QH=192), VD=128, LORA=512.
// Data movement: GLOBAL_LOAD_ASYNC_TO_LDS_B128 (ASYNCcnt) for tile staging,
// transposed LDS layouts so every WMMA operand is fed by ds_load_b128.
// ---------------------------------------------------------------------------

#define B_  2
#define S_  2048
#define E_  2048
#define H_  16
#define NOPE_ 128
#define ROPE_ 64
#define QH_  192
#define VD_  128
#define LORA_ 512
#define M_  (B_ * S_)            // 4096 token rows

typedef unsigned short u16;
typedef unsigned int   u32;

typedef __attribute__((ext_vector_type(16))) __bf16 v16bf;
typedef __attribute__((ext_vector_type(8)))  float  v8f;

union Frag {                 // one WMMA 16-bit operand: 8 VGPRs / lane
    v16bf v;
    u16   u[16];
    u32   d[8];
};

static __device__ inline u16 f32_to_bf16(float f) {
    u32 x = __builtin_bit_cast(u32, f);
    u32 r = (x + 0x7FFFu + ((x >> 16) & 1u)) >> 16;   // round-to-nearest-even
    return (u16)r;
}
static __device__ inline float bf16_to_f32(u16 h) {
    u32 x = ((u32)h) << 16;
    return __builtin_bit_cast(float, x);
}
static __device__ inline v8f vzero8() {
    v8f r;
    #pragma unroll
    for (int i = 0; i < 8; ++i) r[i] = 0.0f;
    return r;
}
static __device__ inline v8f wmma_bf16(v16bf a, v16bf b, v8f c) {
    // D = A(16x32 bf16) * B(32x16 bf16) + C(16x16 f32)
    return __builtin_amdgcn_wmma_f32_16x16x32_bf16(
        false, a, false, b, (short)0, c, false, false);
}

// --- CDNA5 async global->LDS (tracked by ASYNCcnt), per-lane addresses ------
static __device__ inline void async_load_b128(u32 lds_off, const void* gaddr) {
    asm volatile("global_load_async_to_lds_b128 %0, %1, off"
                 :: "v"(lds_off), "v"(gaddr)
                 : "memory");
}
static __device__ inline void wait_asynccnt0() {
    asm volatile("s_wait_asynccnt 0x0" ::: "memory");
}

// ---------------------------------------------------------------------------
// f32 -> bf16 cast
// ---------------------------------------------------------------------------
__global__ __launch_bounds__(256)
void cast_f32_to_bf16(const float* __restrict__ in, u16* __restrict__ out, long n) {
    long i = (long)blockIdx.x * 256 + threadIdx.x;
    if (i < n) out[i] = f32_to_bf16(in[i]);
}

// ---------------------------------------------------------------------------
// Generic bf16 GEMM: C[M,N] = A[M,K] * B[K,N]; row-major; M%128==0, K%32==0,
// N%8==0 (bounds-checked per 8-wide chunk / per column on store).
// Block = 256 threads (8 wave32), tile 128x128x32; wave tile 64(M) x 32(N).
// Double-buffered LDS: A staged via async-to-LDS, B staged via registers with
// a transpose store (BsT[n][k]) so B fragments read as contiguous b128.
// ---------------------------------------------------------------------------
union BChunk { uint4 q; u16 u[8]; };

template <int STORE_BF16>
__global__ __launch_bounds__(256)
void gemm_bf16_wmma(const u16* __restrict__ A, const u16* __restrict__ Bm,
                    void* __restrict__ C, int M, int N, int K) {
    __shared__ u16 As [2][128][40];    // [buf][m][k]  (32 + pad)
    __shared__ u16 BsT[2][128][40];    // [buf][n][k]  (transposed B)

    const int tid   = threadIdx.x;
    const int lane  = tid & 31;
    const int wave  = tid >> 5;
    const int waveM = wave >> 2;          // 0..1
    const int waveN = wave & 3;           // 0..3
    const int lo    = lane & 15;
    const int hi    = lane >> 4;
    const int mBlk  = blockIdx.y * 128;
    const int nBlk  = blockIdx.x * 128;

    auto asyncA = [&](int k0, int buf) {
        #pragma unroll
        for (int c = 0; c < 2; ++c) {
            int idx  = tid + c * 256;           // 512 chunks of 8 over 128x32
            int row  = idx >> 2;
            int col8 = (idx & 3) * 8;
            u32 lds_off = (u32)(size_t)&As[buf][row][col8];
            async_load_b128(lds_off, A + (size_t)(mBlk + row) * K + k0 + col8);
        }
    };
    auto loadB = [&](int k0, BChunk bc[2]) {
        #pragma unroll
        for (int c = 0; c < 2; ++c) {
            int idx  = tid + c * 256;           // 512 chunks of 8 over 32x128
            int row  = idx >> 4;
            int col8 = (idx & 15) * 8;
            int gc   = nBlk + col8;
            if (gc + 8 <= N) {
                bc[c].q = *(const uint4*)(Bm + (size_t)(k0 + row) * N + gc);
            } else {
                bc[c].q.x = bc[c].q.y = bc[c].q.z = bc[c].q.w = 0u;
            }
        }
    };
    auto storeBT = [&](int buf, BChunk bc[2]) {
        #pragma unroll
        for (int c = 0; c < 2; ++c) {
            int idx  = tid + c * 256;
            int row  = idx >> 4;                // k within tile
            int col8 = (idx & 15) * 8;          // n within tile
            #pragma unroll
            for (int j = 0; j < 8; ++j)
                BsT[buf][col8 + j][row] = bc[c].u[j];
        }
    };

    v8f acc[4][2];
    #pragma unroll
    for (int mt = 0; mt < 4; ++mt)
        #pragma unroll
        for (int nt = 0; nt < 2; ++nt) acc[mt][nt] = vzero8();

    const int nk = K >> 5;
    BChunk breg[2];

    // prologue: stage tile 0
    asyncA(0, 0);
    loadB(0, breg);
    storeBT(0, breg);
    wait_asynccnt0();
    __syncthreads();

    for (int kt = 0; kt < nk; ++kt) {
        const int  cur  = kt & 1;
        const int  nxt  = cur ^ 1;
        const bool more = (kt + 1) < nk;

        if (more) {                      // issue next tile early (latency hide)
            asyncA((kt + 1) << 5, nxt);
            loadB((kt + 1) << 5, breg);
        }

        // ---- fragments from current buffers ----
        Frag a[4], b[2];
        #pragma unroll
        for (int nt = 0; nt < 2; ++nt) {
            int cb = waveN * 32 + nt * 16 + lo;
            int kb = hi * 16;
            #pragma unroll
            for (int i = 0; i < 8; ++i)   // 16 contiguous u16 -> 2x ds_load_b128
                b[nt].d[i] = *(const u32*)&BsT[cur][cb][kb + 2 * i];
        }
        #pragma unroll
        for (int mt = 0; mt < 4; ++mt) {
            int rb = waveM * 64 + mt * 16 + lo;
            int kh = hi * 16;
            #pragma unroll
            for (int i = 0; i < 8; ++i) {
                a[mt].d[i] = *(const u32*)&As[cur][rb][kh + 2 * i];
            }
        }
        #pragma unroll
        for (int mt = 0; mt < 4; ++mt)
            #pragma unroll
            for (int nt = 0; nt < 2; ++nt)
                acc[mt][nt] = wmma_bf16(a[mt].v, b[nt].v, acc[mt][nt]);

        if (more) storeBT(nxt, breg);
        wait_asynccnt0();                // our async writes to nxt landed
        __syncthreads();                 // everyone done reading cur / writing nxt
    }

    // ---- epilogue ----
    #pragma unroll
    for (int mt = 0; mt < 4; ++mt) {
        #pragma unroll
        for (int nt = 0; nt < 2; ++nt) {
            int col = nBlk + waveN * 32 + nt * 16 + lo;
            if (col < N) {
                #pragma unroll
                for (int r = 0; r < 8; ++r) {
                    int row = mBlk + waveM * 64 + mt * 16 + r + hi * 8;
                    if (STORE_BF16)
                        ((u16*)C)[(size_t)row * N + col] = f32_to_bf16(acc[mt][nt][r]);
                    else
                        ((float*)C)[(size_t)row * N + col] = acc[mt][nt][r];
                }
            }
        }
    }
}

// ---------------------------------------------------------------------------
// RMSNorm over ckv (first 512 of kva row) * kv_norm_scale -> bf16
// one block (256 thr) per token row
// ---------------------------------------------------------------------------
__global__ __launch_bounds__(256)
void rmsnorm_ckv(const float* __restrict__ kva, const float* __restrict__ gamma,
                 u16* __restrict__ ckv_bf) {
    __shared__ float red[8];
    const int row = blockIdx.x;
    const int tid = threadIdx.x;
    const float* src = kva + (size_t)row * (LORA_ + ROPE_);
    float v0 = src[tid];
    float v1 = src[tid + 256];
    float ss = v0 * v0 + v1 * v1;
    #pragma unroll
    for (int off = 16; off >= 1; off >>= 1) ss += __shfl_xor(ss, off, 32);
    if ((tid & 31) == 0) red[tid >> 5] = ss;
    __syncthreads();
    float sum = 0.f;
    #pragma unroll
    for (int i = 0; i < 8; ++i) sum += red[i];
    float rms = rsqrtf(sum / (float)LORA_ + 1e-6f);
    ckv_bf[(size_t)row * LORA_ + tid]       = f32_to_bf16(v0 * rms * gamma[tid]);
    ckv_bf[(size_t)row * LORA_ + tid + 256] = f32_to_bf16(v1 * rms * gamma[tid + 256]);
}

// ---------------------------------------------------------------------------
// RoPE helpers: pair j angle = pos * 10000^(-j/32), j in [0,32)
// ---------------------------------------------------------------------------
static __device__ inline void rope_cs(int pos, int j, float& c, float& s) {
    float freq = __powf(10000.0f, -((float)j) / 32.0f);
    float ang  = (float)pos * freq;
    c = __cosf(ang);
    s = __sinf(ang);
}

// q_raw [B,S,H,192] bf16  ->  qT [B,H,S,192] bf16 with RoPE on last 64
__global__ __launch_bounds__(192)
void repack_rope_q(const u16* __restrict__ qraw, u16* __restrict__ qT) {
    const int s = blockIdx.x, h = blockIdx.y, b = blockIdx.z;
    const int d = threadIdx.x;
    const u16* src = qraw + (((size_t)(b * S_ + s) * H_) + h) * QH_;
    u16* dst = qT + ((((size_t)b * H_ + h) * S_) + s) * QH_;
    if (d < NOPE_) {
        dst[d] = src[d];
    } else {
        int j = (d - NOPE_) & 31;
        bool hi2 = (d - NOPE_) >= 32;
        float x1 = bf16_to_f32(src[NOPE_ + j]);
        float x2 = bf16_to_f32(src[NOPE_ + 32 + j]);
        float c, sn; rope_cs(s, j, c, sn);
        float out = hi2 ? (x2 * c + x1 * sn) : (x1 * c - x2 * sn);
        dst[d] = f32_to_bf16(out);
    }
}

// kv_raw [B,S,H,256] bf16 + kva f32 (k_pe tail) ->
//   kT [B,H,S,192]  and  vT [B,H,VD,S]  (V transposed for b128 LDS fragments)
__global__ __launch_bounds__(256)
void repack_rope_kv(const u16* __restrict__ kvraw, const float* __restrict__ kva,
                    u16* __restrict__ kT, u16* __restrict__ vT) {
    const int s = blockIdx.x, h = blockIdx.y, b = blockIdx.z;
    const int t = threadIdx.x;
    const u16* src = kvraw + (((size_t)(b * S_ + s) * H_) + h) * (NOPE_ + VD_);
    u16* kdst = kT + ((((size_t)b * H_ + h) * S_) + s) * QH_;
    u16* vdst = vT + (((size_t)b * H_ + h) * VD_) * S_;
    if (t < NOPE_) {
        kdst[t] = src[t];                       // k_nope
        vdst[(size_t)t * S_ + s] = src[NOPE_ + t];  // v, transposed [d][s]
    } else if (t < QH_) {
        int j = (t - NOPE_) & 31;
        bool hi2 = (t - NOPE_) >= 32;
        const float* pe = kva + (size_t)(b * S_ + s) * (LORA_ + ROPE_) + LORA_;
        float x1 = pe[j], x2 = pe[32 + j];
        float c, sn; rope_cs(s, j, c, sn);
        float out = hi2 ? (x2 * c + x1 * sn) : (x1 * c - x2 * sn);
        kdst[t] = f32_to_bf16(out);
    }
}

// ---------------------------------------------------------------------------
// Flash attention (causal). Block = 256 thr = 8 waves; 128 queries per block,
// each wave owns 16 query rows; 64-key tiles staged via async-to-LDS.
// grid = (S/128, H, B)
// ---------------------------------------------------------------------------
__global__ __launch_bounds__(256)
void mla_flash_attention(const u16* __restrict__ qT, const u16* __restrict__ kT,
                         const u16* __restrict__ vT, u16* __restrict__ attnOut) {
    __shared__ u16 Ks [64][200];      // [key][d]  192 + pad
    __shared__ u16 VsT[128][72];      // [d][key]   64 + pad (V pre-transposed)
    __shared__ u16 Ps [8][16][72];    // per-wave P scratch (64 + pad)

    const int tid  = threadIdx.x;
    const int lane = tid & 31;
    const int wave = tid >> 5;
    const int lo   = lane & 15;
    const int hi   = lane >> 4;
    const int h    = blockIdx.y;
    const int b    = blockIdx.z;
    const int qTile = blockIdx.x * 128;
    const int qRow0 = qTile + wave * 16;

    const u16* qB = qT + (((size_t)b * H_ + h) * S_) * QH_;
    const u16* kB = kT + (((size_t)b * H_ + h) * S_) * QH_;
    const u16* vB = vT + (((size_t)b * H_ + h) * VD_) * S_;

    // --- preload Q fragments (6 k-slabs of 32 over d=192) ---
    Frag aq[6];
    {
        const u16* qrow = qB + (size_t)(qRow0 + lo) * QH_;
        int kh = hi * 16;
        #pragma unroll
        for (int ks = 0; ks < 6; ++ks) {
            #pragma unroll
            for (int i = 0; i < 8; ++i) {
                aq[ks].d[i] = *(const u32*)(qrow + ks * 32 + kh + 2 * i);
            }
        }
    }

    v8f o[8];
    #pragma unroll
    for (int dt = 0; dt < 8; ++dt) o[dt] = vzero8();
    float mrow[8], lrow[8];
    #pragma unroll
    for (int r = 0; r < 8; ++r) { mrow[r] = -3.0e38f; lrow[r] = 0.0f; }

    const float scale = 0.07216878365f;   // 192^-0.5
    const int nkt = 2 * blockIdx.x + 2;   // causal: tiles covering keys <= qTile+127

    for (int kt = 0; kt < nkt; ++kt) {
        const int keyBase = kt * 64;
        // ---- async K/V tile staging ----
        {
            int row = tid >> 2, part = tid & 3;          // K: 4 thr per key row
            const u16* krow = kB + (size_t)(keyBase + row) * QH_;
            #pragma unroll
            for (int c = 0; c < 6; ++c) {
                int col8 = part * 48 + c * 8;
                async_load_b128((u32)(size_t)&Ks[row][col8], krow + col8);
            }
            int d = tid >> 1, p2 = tid & 1;              // V^T: 2 thr per d row
            const u16* vrow = vB + (size_t)d * S_ + keyBase;
            #pragma unroll
            for (int c = 0; c < 4; ++c) {
                int col8 = p2 * 32 + c * 8;
                async_load_b128((u32)(size_t)&VsT[d][col8], vrow + col8);
            }
        }
        wait_asynccnt0();
        __syncthreads();

        // ---- S = Q K^T : 4 n-tiles x 6 k-slabs ----
        v8f s[4];
        #pragma unroll
        for (int nt = 0; nt < 4; ++nt) {
            s[nt] = vzero8();
            #pragma unroll
            for (int ks = 0; ks < 6; ++ks) {
                Frag bk;
                int krow = nt * 16 + lo;       // lane = key (N) index
                int kb   = hi * 16;            // d-offset half
                #pragma unroll
                for (int i = 0; i < 8; ++i)    // pairs contiguous along d
                    bk.d[i] = *(const u32*)&Ks[krow][ks * 32 + kb + 2 * i];
                s[nt] = wmma_bf16(aq[ks].v, bk.v, s[nt]);
            }
        }

        // ---- online softmax (row stats per-lane, replicated in 16-lane halves) ----
        float rmax[8];
        #pragma unroll
        for (int r = 0; r < 8; ++r) {
            int qIdx = qRow0 + r + hi * 8;
            float mx = -3.0e38f;
            #pragma unroll
            for (int nt = 0; nt < 4; ++nt) {
                int kIdx = keyBase + nt * 16 + lo;
                float v = s[nt][r] * scale;
                if (kIdx > qIdx) v = -3.0e38f;
                s[nt][r] = v;
                mx = fmaxf(mx, v);
            }
            #pragma unroll
            for (int off = 1; off < 16; off <<= 1)
                mx = fmaxf(mx, __shfl_xor(mx, off, 32));
            rmax[r] = mx;
        }
        #pragma unroll
        for (int r = 0; r < 8; ++r) {
            float mnew = fmaxf(mrow[r], rmax[r]);
            float corr = __expf(mrow[r] - mnew);
            mrow[r] = mnew;
            float rsum = 0.0f;
            #pragma unroll
            for (int nt = 0; nt < 4; ++nt) {
                float p = __expf(s[nt][r] - mnew);
                rsum += p;
                Ps[wave][r + hi * 8][nt * 16 + lo] = f32_to_bf16(p);
            }
            #pragma unroll
            for (int off = 1; off < 16; off <<= 1)
                rsum += __shfl_xor(rsum, off, 32);
            lrow[r] = lrow[r] * corr + rsum;
            #pragma unroll
            for (int dt = 0; dt < 8; ++dt) o[dt][r] *= corr;
        }

        // ---- O += P V : 2 k-slabs of 32 over the 64 keys ----
        #pragma unroll
        for (int ks2 = 0; ks2 < 2; ++ks2) {
            Frag ap;
            int kh = hi * 16;
            #pragma unroll
            for (int i = 0; i < 8; ++i) {
                ap.d[i] = *(const u32*)&Ps[wave][lo][ks2 * 32 + kh + 2 * i];
            }
            #pragma unroll
            for (int dt = 0; dt < 8; ++dt) {
                Frag bv;
                int dcol = dt * 16 + lo;       // lane = output d column
                int kb   = hi * 16;            // key-offset half
                #pragma unroll
                for (int i = 0; i < 8; ++i)    // contiguous keys in VsT row
                    bv.d[i] = *(const u32*)&VsT[dcol][ks2 * 32 + kb + 2 * i];
                o[dt] = wmma_bf16(ap.v, bv.v, o[dt]);
            }
        }
        __syncthreads();
    }

    // ---- finalize & store bf16 to [B,S,H*128] ----
    #pragma unroll
    for (int dt = 0; dt < 8; ++dt) {
        #pragma unroll
        for (int r = 0; r < 8; ++r) {
            int qIdx = qRow0 + r + hi * 8;
            float val = o[dt][r] / lrow[r];
            attnOut[((size_t)b * S_ + qIdx) * (H_ * VD_) + h * VD_ + dt * 16 + lo] =
                f32_to_bf16(val);
        }
    }
}

// ---------------------------------------------------------------------------
// Host launcher
// ---------------------------------------------------------------------------
extern "C" void kernel_launch(void* const* d_in, const int* in_sizes, int n_in,
                              void* d_out, int out_size, void* d_ws, size_t ws_size,
                              hipStream_t stream) {
    (void)in_sizes; (void)n_in; (void)out_size; (void)ws_size;
    const float* x      = (const float*)d_in[0];
    const float* wq     = (const float*)d_in[1];
    const float* wkv_a  = (const float*)d_in[2];
    const float* wkv_b  = (const float*)d_in[3];
    const float* wo     = (const float*)d_in[4];
    const float* gamma  = (const float*)d_in[5];
    float* out = (float*)d_out;

    char* ws = (char*)d_ws;
    size_t off = 0;
    auto alloc = [&](size_t bytes) -> void* {
        void* p = ws + off;
        off += (bytes + 255) & ~(size_t)255;
        return p;
    };
    u16*   x_bf    = (u16*)alloc((size_t)M_ * E_ * 2);
    u16*   wq_bf   = (u16*)alloc((size_t)E_ * (H_ * QH_) * 2);
    u16*   wkva_bf = (u16*)alloc((size_t)E_ * (LORA_ + ROPE_) * 2);
    u16*   wkvb_bf = (u16*)alloc((size_t)LORA_ * (H_ * (NOPE_ + VD_)) * 2);
    u16*   wo_bf   = (u16*)alloc((size_t)(H_ * VD_) * E_ * 2);
    u16*   q_raw   = (u16*)alloc((size_t)M_ * H_ * QH_ * 2);
    float* kva_f   = (float*)alloc((size_t)M_ * (LORA_ + ROPE_) * 4);
    u16*   ckv_bf  = (u16*)alloc((size_t)M_ * LORA_ * 2);
    u16*   kv_raw  = (u16*)alloc((size_t)M_ * H_ * (NOPE_ + VD_) * 2);
    u16*   qT      = (u16*)alloc((size_t)M_ * H_ * QH_ * 2);
    u16*   kTb     = (u16*)alloc((size_t)M_ * H_ * QH_ * 2);
    u16*   vTb     = (u16*)alloc((size_t)M_ * H_ * VD_ * 2);
    u16*   attn_bf = (u16*)alloc((size_t)M_ * (H_ * VD_) * 2);

    auto cast = [&](const float* src, u16* dst, long n) {
        cast_f32_to_bf16<<<dim3((unsigned)((n + 255) / 256)), dim3(256), 0, stream>>>(src, dst, n);
    };
    cast(x,     x_bf,    (long)M_ * E_);
    cast(wq,    wq_bf,   (long)E_ * (H_ * QH_));
    cast(wkv_a, wkva_bf, (long)E_ * (LORA_ + ROPE_));
    cast(wkv_b, wkvb_bf, (long)LORA_ * (H_ * (NOPE_ + VD_)));
    cast(wo,    wo_bf,   (long)(H_ * VD_) * E_);

    // q = x @ wq   (4096 x 3072, K=2048) -> bf16
    gemm_bf16_wmma<1><<<dim3((H_ * QH_ + 127) / 128, M_ / 128), dim3(256), 0, stream>>>(
        x_bf, wq_bf, (void*)q_raw, M_, H_ * QH_, E_);
    // kva = x @ wkv_a  (4096 x 576, K=2048) -> f32 (needed for RMSNorm + RoPE precision)
    gemm_bf16_wmma<0><<<dim3((LORA_ + ROPE_ + 127) / 128, M_ / 128), dim3(256), 0, stream>>>(
        x_bf, wkva_bf, (void*)kva_f, M_, LORA_ + ROPE_, E_);
    // RMSNorm(ckv) * gamma -> bf16
    rmsnorm_ckv<<<dim3(M_), dim3(256), 0, stream>>>(kva_f, gamma, ckv_bf);
    // kv = ckv @ wkv_b  (4096 x 4096, K=512) -> bf16
    gemm_bf16_wmma<1><<<dim3((H_ * (NOPE_ + VD_) + 127) / 128, M_ / 128), dim3(256), 0, stream>>>(
        ckv_bf, wkvb_bf, (void*)kv_raw, M_, H_ * (NOPE_ + VD_), LORA_);
    // RoPE + layout repack
    repack_rope_q<<<dim3(S_, H_, B_), dim3(QH_), 0, stream>>>(q_raw, qT);
    repack_rope_kv<<<dim3(S_, H_, B_), dim3(256), 0, stream>>>(kv_raw, kva_f, kTb, vTb);
    // causal flash attention
    mla_flash_attention<<<dim3(S_ / 128, H_, B_), dim3(256), 0, stream>>>(qT, kTb, vTb, attn_bf);
    // out = attn @ wo  (4096 x 2048, K=2048) -> f32 to d_out
    gemm_bf16_wmma<0><<<dim3((E_ + 127) / 128, M_ / 128), dim3(256), 0, stream>>>(
        attn_bf, wo_bf, (void*)out, M_, E_, H_ * VD_);
}
